// FPScaledLinear_55130200212236
// MI455X (gfx1250) — matrix operand order repaired and checked by reading, not compiled, not run on hardware
//
#include <hip/hip_runtime.h>

typedef __attribute__((ext_vector_type(16))) __bf16 v16bf;
typedef __attribute__((ext_vector_type(8)))  __bf16 v8bf;
typedef __attribute__((ext_vector_type(4)))  __bf16 v4bf;
typedef __attribute__((ext_vector_type(8)))  float  v8f;
typedef __attribute__((ext_vector_type(4)))  float  v4f;

// Problem constants (B=4, S=4096, D_IN=2048, D_OUT=2048)
constexpr int Mdim = 4 * 4096;   // 16384 rows of x
constexpr int Ndim = 2048;       // out features
constexpr int Kdim = 2048;       // in features
// Tiling
constexpr int BM = 64;           // workgroup M tile
constexpr int BN = 128;          // workgroup N tile
constexpr int BK = 32;           // K step (one 16x16x32 WMMA deep)
constexpr int SK = BK + 8;       // padded LDS row stride (bf16 elems) -> 80B rows
constexpr int NSTEP = Kdim / BK; // 64

__global__ __launch_bounds__(256)
void fpscaled_linear_wmma_bf16x3(const float* __restrict__ X,
                                 const float* __restrict__ Wt,
                                 const float* __restrict__ Sc,
                                 const float* __restrict__ Bi,
                                 float* __restrict__ O)
{
    // Double-buffered hi/lo bf16 tiles (total 60 KB)
    __shared__ __bf16 sAh[2][BM * SK];
    __shared__ __bf16 sAl[2][BM * SK];
    __shared__ __bf16 sBh[2][BN * SK];
    __shared__ __bf16 sBl[2][BN * SK];

    const int tid  = threadIdx.x;
    const int lane = tid & 31;
    const int l16  = lane & 15;
    const int hhalf = lane >> 4;          // 0 or 1 (lane >= 16)
    const int wave = tid >> 5;            // 8 waves
    const int wm   = wave >> 2;           // 0..1  -> 32-row slice
    const int wn   = wave & 3;            // 0..3  -> 32-col slice
    const int gm0  = blockIdx.y * BM;
    const int gn0  = blockIdx.x * BN;

    // fp32 staging registers for the in-flight k-tile
    v4f ga[2];   // A: 64x32 f32 = 512 float4 / 256 thr = 2 each
    v4f gb[4];   // B: 128x32 f32 = 1024 float4 / 256 thr = 4 each

    auto loadG = [&](int k0) {
#pragma unroll
        for (int i = 0; i < 2; ++i) {
            int flat = i * 256 + tid;              // 0..511
            int r = flat >> 3, c = (flat & 7) << 2;
            ga[i] = *(const v4f*)(X + (size_t)(gm0 + r) * Kdim + k0 + c);
        }
#pragma unroll
        for (int i = 0; i < 4; ++i) {
            int flat = i * 256 + tid;              // 0..1023
            int r = flat >> 3, c = (flat & 7) << 2;
            gb[i] = *(const v4f*)(Wt + (size_t)(gn0 + r) * Kdim + k0 + c);
        }
    };

    auto splitStore = [&](int buf) {
#pragma unroll
        for (int i = 0; i < 2; ++i) {
            int flat = i * 256 + tid;
            int r = flat >> 3, c = (flat & 7) << 2;
            v4bf h, l;
#pragma unroll
            for (int j = 0; j < 4; ++j) {
                float f = ga[i][j];
                __bf16 hb = (__bf16)f;            // RNE truncate to bf16
                h[j] = hb;
                l[j] = (__bf16)(f - (float)hb);   // residual term
            }
            *(v4bf*)&sAh[buf][r * SK + c] = h;    // ds_store_b64
            *(v4bf*)&sAl[buf][r * SK + c] = l;
        }
#pragma unroll
        for (int i = 0; i < 4; ++i) {
            int flat = i * 256 + tid;
            int r = flat >> 3, c = (flat & 7) << 2;
            v4bf h, l;
#pragma unroll
            for (int j = 0; j < 4; ++j) {
                float f = gb[i][j];
                __bf16 hb = (__bf16)f;
                h[j] = hb;
                l[j] = (__bf16)(f - (float)hb);
            }
            *(v4bf*)&sBh[buf][r * SK + c] = h;
            *(v4bf*)&sBl[buf][r * SK + c] = l;
        }
    };

    // A fragment (16x32, MxK): lane<16 holds K[0..7]+K[16..23], lane>=16 holds
    // K[8..15]+K[24..31] of row M=lane%16  -> two ds_load_b128 per fragment.
    auto fragA = [&](const __bf16* base, int row) -> v16bf {
        const __bf16* p = base + row * SK;
        int kb = hhalf ? 8 : 0;
        v8bf c0 = *(const v8bf*)(p + kb);
        v8bf c1 = *(const v8bf*)(p + kb + 16);
        return __builtin_shufflevector(c0, c1, 0, 1, 2, 3, 4, 5, 6, 7,
                                               8, 9, 10, 11, 12, 13, 14, 15);
    };
    // B fragment (32x16, KxN): lane<16 holds K[0..7]+K[16..23], lane>=16 holds
    // K[8..15]+K[24..31] of column N=lane%16 -> two ds_load_b128.
    auto fragB = [&](const __bf16* base, int row) -> v16bf {
        const __bf16* p = base + row * SK;
        int kb = hhalf ? 8 : 0;
        v8bf c0 = *(const v8bf*)(p + kb);
        v8bf c1 = *(const v8bf*)(p + kb + 16);
        return __builtin_shufflevector(c0, c1, 0, 1, 2, 3, 4, 5, 6, 7,
                                               8, 9, 10, 11, 12, 13, 14, 15);
    };

    v8f acc[2][2];
#pragma unroll
    for (int mi = 0; mi < 2; ++mi)
#pragma unroll
        for (int ni = 0; ni < 2; ++ni)
            acc[mi][ni] = (v8f){};

    // Prologue: fill buffer 0
    loadG(0);
    splitStore(0);

#pragma unroll 2
    for (int ks = 0; ks < NSTEP; ++ks) {
        __syncthreads();
        const int cur = ks & 1;

        if (ks + 1 < NSTEP) loadG((ks + 1) * BK);      // hide HBM latency
        if (ks + 2 < NSTEP) {                          // L2 prefetch 2 ahead
            int k2 = (ks + 2) * BK;
            __builtin_prefetch(X  + (size_t)(gm0 + (tid >> 3)) * Kdim + k2 + ((tid & 7) << 2), 0, 1);
            __builtin_prefetch(Wt + (size_t)(gn0 + (tid >> 2)) * Kdim + k2 + ((tid & 3) << 3), 0, 1);
        }

        // A fragments for this wave's two 16-row tiles (hi + lo)
        v16bf ah[2], al[2];
#pragma unroll
        for (int mi = 0; mi < 2; ++mi) {
            int row = wm * 32 + mi * 16 + l16;
            ah[mi] = fragA(sAh[cur], row);
            al[mi] = fragA(sAl[cur], row);
        }

#pragma unroll
        for (int ni = 0; ni < 2; ++ni) {
            int row = wn * 32 + ni * 16 + l16;
            v16bf bh = fragB(sBh[cur], row);
            v16bf bl = fragB(sBl[cur], row);
#pragma unroll
            for (int mi = 0; mi < 2; ++mi) {
                // bf16x3 fp32 emulation: hi*hi + lo*hi + hi*lo
                acc[mi][ni] = __builtin_amdgcn_wmma_f32_16x16x32_bf16(
                    false, ah[mi], false, bh, (short)0, acc[mi][ni], false, false);
                acc[mi][ni] = __builtin_amdgcn_wmma_f32_16x16x32_bf16(
                    false, al[mi], false, bh, (short)0, acc[mi][ni], false, false);
                acc[mi][ni] = __builtin_amdgcn_wmma_f32_16x16x32_bf16(
                    false, ah[mi], false, bl, (short)0, acc[mi][ni], false, false);
            }
        }

        if (ks + 1 < NSTEP) splitStore((ks + 1) & 1);
    }

    // Epilogue: D layout = VGPR i -> row (i + 8*(lane>=16)), col = lane%16.
    // Apply per-out-feature scale & bias, store f32.
#pragma unroll
    for (int ni = 0; ni < 2; ++ni) {
        int col = gn0 + wn * 32 + ni * 16 + l16;
        float sc = Sc[col];
        float bs = Bi[col];
#pragma unroll
        for (int mi = 0; mi < 2; ++mi) {
            int rbase = gm0 + wm * 32 + mi * 16 + hhalf * 8;
#pragma unroll
            for (int i = 0; i < 8; ++i) {
                O[(size_t)(rbase + i) * Ndim + col] = acc[mi][ni][i] * sc + bs;
            }
        }
    }
}

extern "C" void kernel_launch(void* const* d_in, const int* in_sizes, int n_in,
                              void* d_out, int out_size, void* d_ws, size_t ws_size,
                              hipStream_t stream) {
    (void)in_sizes; (void)n_in; (void)out_size; (void)d_ws; (void)ws_size;
    const float* X  = (const float*)d_in[0];   // [4,4096,2048]
    const float* Wt = (const float*)d_in[1];   // [2048,2048] (out, in)
    const float* Sc = (const float*)d_in[2];   // [2048]
    const float* Bi = (const float*)d_in[3];   // [2048]
    float* O = (float*)d_out;                  // [4,4096,2048]

    dim3 grid(Ndim / BN, Mdim / BM);           // (16, 256)
    fpscaled_linear_wmma_bf16x3<<<grid, 256, 0, stream>>>(X, Wt, Sc, Bi, O);
}
